// ShowAttendTellCore_54030688584114
// MI455X (gfx1250) — matrix operand order repaired and checked by reading, not compiled, not run on hardware
//
#include <hip/hip_runtime.h>
#include <hip/hip_bf16.h>
#include <math.h>

// ---------------------------------------------------------------------------
// ShowAttendTell core for MI455X (gfx1250, wave32, WMMA).
// Big GEMMs run as bf16 V_WMMA_F32_16X16X32_BF16 with f32 accumulation.
// ---------------------------------------------------------------------------

typedef __attribute__((ext_vector_type(16))) __bf16 v16bf;
typedef __attribute__((ext_vector_type(8)))  float  v8f;
typedef __attribute__((ext_vector_type(4)))  float  v4f;
typedef __attribute__((ext_vector_type(2)))  float  v2f;
typedef __attribute__((ext_vector_type(4)))  unsigned int v4u;
typedef __attribute__((ext_vector_type(2)))  unsigned int v2u;
typedef unsigned short u16;

union BF16x16 {           // one WMMA 16-half operand fragment
  v16bf  v;
  __bf16 h[16];
  u16    u[16];
  unsigned int w[8];
  v4u    q[2];
};

// f32 -> bf16 via native hardware convert (clang lowers to v_cvt_*bf16*)
__device__ __forceinline__ u16 f2bf(float f) {
  __bf16 b = (__bf16)f;
  return __builtin_bit_cast(u16, b);
}
__device__ __forceinline__ float sigm(float x) { return 1.0f / (1.0f + expf(-x)); }

// A-matrix (16x32 bf16, MxK) K-pair base for VGPR v, lane-half hi
// (per CDNA5 ISA 7.12.2: lanes0-15 v0..3 -> K0..7, v4..7 -> K16..23;
//  lanes16-31 v0..3 -> K8..15, v4..7 -> K24..31)
__device__ __forceinline__ int a_kpair(int v, int hi) {
  return (v < 4) ? (2 * v + 8 * hi) : (16 + 2 * (v - 4) + 8 * hi);
}

#define BB   512
#define LL   196
#define DATT 2048
#define HATT 512
#define DRNN 512
#define KCAT 3072            // 512 (xt) + 2048 (att_res) + 512 (h_last)
#define MROWS (BB * LL)      // 100352

// ---------------------------------------------------------------------------
// Prep 1: att_h[b][h] = h_last[b] . W_h2att[h] + b_h2att[h]   (tiny GEMM)
// ---------------------------------------------------------------------------
__global__ void k_atth(const float* __restrict__ h_last,
                       const float* __restrict__ W_h2att,
                       const float* __restrict__ b_h2att,
                       float* __restrict__ atth) {
  const int b = blockIdx.x;
  __shared__ float hl[DRNN];
  for (int i = threadIdx.x; i < DRNN; i += 256) hl[i] = h_last[(size_t)b * DRNN + i];
  __syncthreads();
  for (int hh = threadIdx.x; hh < HATT; hh += 256) {
    const float* wr = W_h2att + (size_t)hh * DRNN;
    float acc = b_h2att[hh];
    #pragma unroll 8
    for (int d = 0; d < DRNN; ++d) acc = fmaf(hl[d], wr[d], acc);
    atth[(size_t)b * HATT + hh] = acc;
  }
}

// ---------------------------------------------------------------------------
// Prep 2: wB[d][h] = bf16(W_ctx[h][d])   — transposed to K-major for staging
// ---------------------------------------------------------------------------
__global__ void k_conv_wctx(const float* __restrict__ W_ctx, u16* __restrict__ wB) {
  int i = blockIdx.x * 256 + threadIdx.x;          // i = d*512 + h
  if (i >= DATT * HATT) return;
  int d = i >> 9, hh = i & 511;
  wB[(size_t)d * HATT + hh] = f2bf(W_ctx[(size_t)hh * DATT + d]);
}

// ---------------------------------------------------------------------------
// Prep 3: Wcat[k][n] = bf16( k<2560 ? W_ih[n][k] : W_hh[n][k-2560] )
// ---------------------------------------------------------------------------
__global__ void k_conv_wcat(const float* __restrict__ W_ih,
                            const float* __restrict__ W_hh,
                            u16* __restrict__ Wcat) {
  int i = blockIdx.x * 256 + threadIdx.x;          // i = k*2048 + n
  if (i >= KCAT * 2048) return;
  int k = i >> 11, n = i & 2047;
  float v = (k < 2560) ? W_ih[(size_t)n * 2560 + k]
                       : W_hh[(size_t)n * DRNN + (k - 2560)];
  Wcat[(size_t)k * 2048 + n] = f2bf(v);
}

// ---------------------------------------------------------------------------
// Prep 4: Xcat static parts: cols [0,512) = xt, cols [2560,3072) = h_last
// ---------------------------------------------------------------------------
__global__ void k_xcat_static(const float* __restrict__ xt,
                              const float* __restrict__ h_last,
                              u16* __restrict__ Xcat) {
  int i = blockIdx.x * 256 + threadIdx.x;          // i = b*1024 + j
  if (i >= BB * 1024) return;
  int b = i >> 10, j = i & 1023;
  float v; int k;
  if (j < 512) { k = j;              v = xt[(size_t)b * 512 + j]; }
  else         { k = 2560 + j - 512; v = h_last[(size_t)b * 512 + (j - 512)]; }
  Xcat[(size_t)b * KCAT + k] = f2bf(v);
}

// ---------------------------------------------------------------------------
// Kernel 2: the big fused attention-score GEMM.
//   scores[bl] = W_alpha . tanh(att_feats[bl] @ W_ctx^T + b_ctx + att_h[b]) + b_alpha
// Block = 8 waves: 4 M-tiles x 2 N-halves (each wave: 16 rows x 256 cols ->
// 16 WMMA accumulators = 128 VGPRs). K loop stages bf16 B (32x512) in LDS.
// B fragments are double-buffered so each WMMA waits on a load issued one
// WMMA earlier (s_wait_dscnt 2) instead of one issued immediately before.
// ---------------------------------------------------------------------------
#define LDSB_STRIDE 520       // halves; 1040B rows -> 16B aligned, banks spread

__global__ void __launch_bounds__(256, 2)
k_attn_scores(const float* __restrict__ att,     // [100352][2048] f32
              const u16*   __restrict__ wB,      // [2048][512] bf16 (K-major)
              const float* __restrict__ atth,    // [512][512]
              const float* __restrict__ b_ctx,   // [512]
              const float* __restrict__ w_alpha, // [512]
              const float* __restrict__ b_alpha, // [1]
              float* __restrict__ scores) {      // [100352]
  __shared__ u16   sB[32 * LDSB_STRIDE];
  __shared__ float sScore[64 * 2];

  const int tid  = threadIdx.x;
  const int lane = tid & 31;
  const int wave = tid >> 5;
  const int mt    = wave >> 1;          // 0..3 : M-tile in block
  const int nhalf = wave & 1;           // 0..1 : N range [nhalf*256, +256)
  const int mcol  = lane & 15;
  const int hi    = lane >> 4;
  const int m0    = blockIdx.x * 64 + mt * 16;
  const int nbase = nhalf * 256;

  v8f acc[16];
  #pragma unroll
  for (int nt = 0; nt < 16; ++nt) acc[nt] = (v8f)(0.0f);

  const float* arow = att + (size_t)(m0 + mcol) * DATT;   // this lane's A row

  for (int k0 = 0; k0 < DATT; k0 += 32) {
    // A pair loads issued before the staging barrier: HBM latency overlaps
    // the cooperative B copy + barrier.
    v2f ap[8];
    #pragma unroll
    for (int v = 0; v < 8; ++v)
      ap[v] = *(const v2f*)(arow + k0 + a_kpair(v, hi));

    __syncthreads();
    // cooperative stage of B chunk [k0..k0+32) x [0..512) bf16 -> LDS
    for (int c = tid; c < 32 * 64; c += 256) {            // 16B chunks
      int r = c >> 6, col = c & 63;
      *(v4u*)&sB[r * LDSB_STRIDE + col * 8] =
          *(const v4u*)(wB + (size_t)(k0 + r) * HATT + col * 8);
    }
    __syncthreads();

    // A fragment: native f32->bf16 converts (ISA 16-bit A layout)
    BF16x16 au;
    #pragma unroll
    for (int v = 0; v < 8; ++v) {
      au.h[2 * v]     = (__bf16)ap[v].x;
      au.h[2 * v + 1] = (__bf16)ap[v].y;
    }

    // 16 N-tiles, B fragment lane = N col, K in A (table) order, double-buffered
    const u16* bcol = &sB[nbase + mcol];
    BF16x16 bu[2];
    #pragma unroll
    for (int s = 0; s < 16; ++s)
      bu[0].u[s] = bcol[(a_kpair(s >> 1, hi) + (s & 1)) * LDSB_STRIDE];
    #pragma unroll
    for (int nt = 0; nt < 16; ++nt) {
      if (nt < 15) {
        #pragma unroll
        for (int s = 0; s < 16; ++s)
          bu[(nt + 1) & 1].u[s] = bcol[(a_kpair(s >> 1, hi) + (s & 1)) * LDSB_STRIDE + (nt + 1) * 16];
      }
      acc[nt] = __builtin_amdgcn_wmma_f32_16x16x32_bf16(
          false, au.v, false, bu[nt & 1].v, (short)0, acc[nt], false, false);
    }
  }

  // epilogue: bias + att_h, tanh, dot with w_alpha (partial per lane)
  float ps[8];
  #pragma unroll
  for (int v = 0; v < 8; ++v) ps[v] = 0.0f;

  #pragma unroll
  for (int nt = 0; nt < 16; ++nt) {
    const int n  = nbase + nt * 16 + mcol;
    const float bc = b_ctx[n];
    const float wa = w_alpha[n];
    #pragma unroll
    for (int v = 0; v < 8; ++v) {
      const int row = m0 + v + 8 * hi;
      const int b   = row / LL;
      float t = tanhf(acc[nt][v] + bc + atth[(size_t)b * HATT + n]);
      ps[v] = fmaf(t, wa, ps[v]);
    }
  }
  // reduce across the 16 lanes sharing each row (stays within half-wave)
  #pragma unroll
  for (int mask = 1; mask <= 8; mask <<= 1) {
    #pragma unroll
    for (int v = 0; v < 8; ++v) ps[v] += __shfl_xor(ps[v], mask, 32);
  }
  if (mcol == 0) {
    #pragma unroll
    for (int v = 0; v < 8; ++v)
      sScore[(mt * 16 + hi * 8 + v) * 2 + nhalf] = ps[v];
  }
  __syncthreads();
  if (tid < 64)
    scores[(size_t)blockIdx.x * 64 + tid] =
        sScore[tid * 2 + 0] + sScore[tid * 2 + 1] + b_alpha[0];
}

// ---------------------------------------------------------------------------
// Kernel 3: softmax over L=196 per batch row
// ---------------------------------------------------------------------------
__global__ void k_softmax(const float* __restrict__ scores, float* __restrict__ wgt) {
  const int b = blockIdx.x, tid = threadIdx.x;
  __shared__ float red[256];
  float s = (tid < LL) ? scores[(size_t)b * LL + tid] : -INFINITY;
  red[tid] = s; __syncthreads();
  for (int off = 128; off > 0; off >>= 1) {
    if (tid < off) red[tid] = fmaxf(red[tid], red[tid + off]);
    __syncthreads();
  }
  const float m = red[0]; __syncthreads();
  float e = (tid < LL) ? expf(s - m) : 0.0f;
  red[tid] = e; __syncthreads();
  for (int off = 128; off > 0; off >>= 1) {
    if (tid < off) red[tid] += red[tid + off];
    __syncthreads();
  }
  if (tid < LL) wgt[(size_t)b * LL + tid] = e / red[0];
}

// ---------------------------------------------------------------------------
// Kernel 4: att_res[b] = sum_l w[b][l] * att_feats[b][l][:]  (bandwidth bound,
// second pass over the 822MB tensor), written straight into Xcat as bf16.
// ---------------------------------------------------------------------------
__global__ void k_attres(const float* __restrict__ att,
                         const float* __restrict__ wgt,
                         u16* __restrict__ Xcat) {
  const int b = blockIdx.x >> 1, ch = blockIdx.x & 1, tid = threadIdx.x;
  __shared__ float w[LL];
  if (tid < LL) w[tid] = wgt[(size_t)b * LL + tid];
  __syncthreads();
  const int d = ch * 1024 + tid * 4;
  const float* base = att + (size_t)b * LL * DATT + d;
  v4f acc = (v4f)(0.0f);
  for (int l = 0; l < LL; ++l) {
    v4f p = *(const v4f*)(base + (size_t)l * DATT);
    acc += w[l] * p;
  }
  v2u pk;
  pk.x = (unsigned)f2bf(acc.x) | ((unsigned)f2bf(acc.y) << 16);
  pk.y = (unsigned)f2bf(acc.z) | ((unsigned)f2bf(acc.w) << 16);
  *(v2u*)(Xcat + (size_t)b * KCAT + 512 + d) = pk;
}

// ---------------------------------------------------------------------------
// Kernel 5: fused gates GEMM + LSTM cell.
//   gates = Xcat @ Wcat  (M=512, N=2048, K=3072, bf16 WMMA)
// Each wave owns one 16x16 state tile and accumulates its i/f/g/o quadrants
// simultaneously, then finishes the LSTM in-register.
// ---------------------------------------------------------------------------
__global__ void __launch_bounds__(256, 4)
k_lstm(const u16* __restrict__ Xcat,    // [512][3072] bf16
       const u16* __restrict__ Wcat,    // [3072][2048] bf16
       const float* __restrict__ c_in,  // [512][512]
       float* __restrict__ out) {       // h_new | h_new | c_new
  const int tid  = threadIdx.x;
  const int lane = tid & 31;
  const int id   = blockIdx.x * 8 + (tid >> 5);   // 1024 tiles
  const int mt = id >> 5, nt = id & 31;
  const int m0 = mt * 16, n0 = nt * 16;
  const int mcol = lane & 15, hi = lane >> 4;

  v8f aI = (v8f)(0.0f), aF = (v8f)(0.0f), aG = (v8f)(0.0f), aO = (v8f)(0.0f);

  const u16* arow = Xcat + (size_t)(m0 + mcol) * KCAT;

  #pragma unroll 2
  for (int k0 = 0; k0 < KCAT; k0 += 32) {
    BF16x16 au;
    #pragma unroll
    for (int v = 0; v < 8; ++v)
      au.w[v] = *(const unsigned int*)(arow + k0 + a_kpair(v, hi));

    const u16* bcol = Wcat + (size_t)k0 * 2048 + n0 + mcol;   // lane = N col
    BF16x16 bI, bF, bG, bO;
    #pragma unroll
    for (int s = 0; s < 16; ++s) {
      const size_t off = (size_t)(a_kpair(s >> 1, hi) + (s & 1)) * 2048;
      bI.u[s] = bcol[off + 0 * 512]; bF.u[s] = bcol[off + 1 * 512];
      bG.u[s] = bcol[off + 2 * 512]; bO.u[s] = bcol[off + 3 * 512];
    }

    aI = __builtin_amdgcn_wmma_f32_16x16x32_bf16(false, au.v, false, bI.v, (short)0, aI, false, false);
    aF = __builtin_amdgcn_wmma_f32_16x16x32_bf16(false, au.v, false, bF.v, (short)0, aF, false, false);
    aG = __builtin_amdgcn_wmma_f32_16x16x32_bf16(false, au.v, false, bG.v, (short)0, aG, false, false);
    aO = __builtin_amdgcn_wmma_f32_16x16x32_bf16(false, au.v, false, bO.v, (short)0, aO, false, false);
  }

  const int S = BB * DRNN;
  const int n = n0 + mcol;
  #pragma unroll
  for (int v = 0; v < 8; ++v) {
    const int m = m0 + v + 8 * hi;
    const float ig = sigm(aI[v]);
    const float fg = sigm(aF[v]);
    const float gg = tanhf(aG[v]);
    const float og = sigm(aO[v]);
    const float cn = fg * c_in[(size_t)m * DRNN + n] + ig * gg;
    const float hn = og * tanhf(cn);
    out[(size_t)m * DRNN + n]         = hn;   // h_new
    out[S + (size_t)m * DRNN + n]     = hn;   // h_new[None]
    out[2 * S + (size_t)m * DRNN + n] = cn;   // c_new[None]
  }
}

// ---------------------------------------------------------------------------
extern "C" void kernel_launch(void* const* d_in, const int* in_sizes, int n_in,
                              void* d_out, int out_size, void* d_ws, size_t ws_size,
                              hipStream_t stream) {
  const float* xt      = (const float*)d_in[0];
  // d_in[1] = fc_feats (unused by reference)
  const float* att     = (const float*)d_in[2];
  const float* h_in    = (const float*)d_in[3];   // [1,B,512]
  const float* c_in    = (const float*)d_in[4];   // [1,B,512]
  const float* W_ctx   = (const float*)d_in[5];
  const float* b_ctx   = (const float*)d_in[6];
  const float* W_h2att = (const float*)d_in[7];
  const float* b_h2att = (const float*)d_in[8];
  const float* W_alpha = (const float*)d_in[9];
  const float* b_alpha = (const float*)d_in[10];
  const float* W_ih    = (const float*)d_in[11];
  const float* W_hh    = (const float*)d_in[12];
  float* out = (float*)d_out;

  char* ws = (char*)d_ws;
  float* atth   = (float*)(ws + 0);           // 1,048,576 B
  float* scores = (float*)(ws + 1048576);     //   401,408 B
  float* wgt    = (float*)(ws + 1449984);     //   401,408 B
  u16*   wB     = (u16*)  (ws + 1851392);     // 2,097,152 B
  u16*   Wcat   = (u16*)  (ws + 3948544);     // 12,582,912 B
  u16*   Xcat   = (u16*)  (ws + 16531456);    // 3,145,728 B  (total ~19.7 MB)

  // prep
  k_atth       <<<BB,                     256, 0, stream>>>(h_in, W_h2att, b_h2att, atth);
  k_conv_wctx  <<<(DATT * HATT) / 256,    256, 0, stream>>>(W_ctx, wB);
  k_conv_wcat  <<<(KCAT * 2048) / 256,    256, 0, stream>>>(W_ih, W_hh, Wcat);
  k_xcat_static<<<(BB * 1024) / 256,      256, 0, stream>>>(xt, h_in, Xcat);
  // attention
  k_attn_scores<<<MROWS / 64,             256, 0, stream>>>(att, wB, atth, b_ctx,
                                                            W_alpha, b_alpha, scores);
  k_softmax    <<<BB,                     256, 0, stream>>>(scores, wgt);
  k_attres     <<<BB * 2,                 256, 0, stream>>>(att, wgt, Xcat);
  // LSTM
  k_lstm       <<<(32 * 32) / 8,          256, 0, stream>>>(Xcat, Wcat, c_in, out);
}
